// TruthGPTSelfAttention_51049981281122
// MI455X (gfx1250) — matrix operand order repaired and checked by reading, not compiled, not run on hardware
//
#include <hip/hip_runtime.h>
#include <hip/hip_bf16.h>

typedef __attribute__((ext_vector_type(16))) __bf16 v16bf;
typedef __attribute__((ext_vector_type(8)))  __bf16 bf16x8;
typedef __attribute__((ext_vector_type(4)))  __bf16 bf16x4;
typedef __attribute__((ext_vector_type(8)))  float  v8f;

static constexpr int Bc  = 2;
static constexpr int Sc  = 2048;
static constexpr int Dc  = 1024;
static constexpr int Hc  = 16;
static constexpr int HDc = 64;
static constexpr int Mg  = Bc * Sc;  // 4096 rows

union FragU { v16bf v; bf16x8 q[2]; };

// ---------------------------------------------------------------------------
// Fragment loaders for v_wmma_f32_16x16x32_bf16 (wave32).
// A (16x32, MxK): lane L<16 -> row L, K = {k..k+7, k+16..k+23}
//                 lane L>=16 -> row L-16, K = {k+8..k+15, k+24..k+31}
// B (32x16, KxN): lane L<16 -> col L, K = {k..k+7, k+16..k+23}
//                 lane L>=16 -> col L-16, K = {k+8..k+15, k+24..k+31}
// Both are loaded from K-contiguous row-major storage (NT GEMM pattern).
// ---------------------------------------------------------------------------
__device__ inline v16bf load_frag_a(const __bf16* base, int ld, int k) {
  const int lane = threadIdx.x & 31;
  const __bf16* p = base + (size_t)(lane & 15) * ld + k + ((lane >> 4) << 3);
  FragU u;
  u.q[0] = *reinterpret_cast<const bf16x8*>(p);
  u.q[1] = *reinterpret_cast<const bf16x8*>(p + 16);
  return u.v;
}

__device__ inline v16bf load_frag_b(const __bf16* base, int ld, int k) {
  const int lane = threadIdx.x & 31;
  const __bf16* p = base + (size_t)(lane & 15) * ld + k + ((lane >> 4) << 3);
  FragU u;
  u.q[0] = *reinterpret_cast<const bf16x8*>(p);
  u.q[1] = *reinterpret_cast<const bf16x8*>(p + 16);
  return u.v;
}

__device__ inline v8f wmma_bf16(v16bf a, v16bf b, v8f c) {
  // (neg_a, A, neg_b, B, c_mod, C, reuse_a, reuse_b)
  return __builtin_amdgcn_wmma_f32_16x16x32_bf16(false, a, false, b, (short)0, c,
                                                 false, false);
}

// ---------------------------------------------------------------------------
// fp32 -> bf16 convert, 4 elements / thread
// ---------------------------------------------------------------------------
__global__ void cvt_f32_bf16(const float* __restrict__ src,
                             __bf16* __restrict__ dst, int n4) {
  const int i = blockIdx.x * blockDim.x + threadIdx.x;
  if (i < n4) {
    const float4 f = reinterpret_cast<const float4*>(src)[i];
    bf16x4 o;
    o[0] = (__bf16)f.x; o[1] = (__bf16)f.y; o[2] = (__bf16)f.z; o[3] = (__bf16)f.w;
    reinterpret_cast<bf16x4*>(dst)[i] = o;
  }
}

// ---------------------------------------------------------------------------
// C[M,N] = A[M,K] x W[N,K]^T + bias.  One wave -> 64x64 output tile
// (4x4 grid of 16x16 WMMA accumulators; 32 FLOP/byte from cache).
// mode 0: bf16 out as [b,h,s,hd]   (Q, K projections)
// mode 2: bf16 out as [b,h,hd,s]   (V projection, transposed for attention)
// mode 3: fp32 out as [m,n]        (final output projection)
// ---------------------------------------------------------------------------
__global__ void gemm_nt(const __bf16* __restrict__ A, const __bf16* __restrict__ W,
                        const float* __restrict__ bias,
                        __bf16* __restrict__ out_bf, float* __restrict__ out_f,
                        int mode) {
  const int lane   = threadIdx.x & 31;
  const int lh     = lane >> 4;
  const int wave   = blockIdx.x * (blockDim.x >> 5) + (threadIdx.x >> 5);
  const int tilesN = Dc / 64;                 // 16
  const int m0     = (wave / tilesN) * 64;
  const int n0     = (wave % tilesN) * 64;
  if (m0 >= Mg) return;

  v8f acc[4][4];
  #pragma unroll
  for (int mi = 0; mi < 4; ++mi)
    #pragma unroll
    for (int ci = 0; ci < 4; ++ci)
      #pragma unroll
      for (int r = 0; r < 8; ++r) acc[mi][ci][r] = 0.f;

  const __bf16* Abase = A + (size_t)m0 * Dc;
  const __bf16* Wbase = W + (size_t)n0 * Dc;

  for (int k = 0; k < Dc; k += 32) {
    // prefetch next K chunk of this lane's A row into cache
    if (k + 32 < Dc)
      __builtin_prefetch(Abase + (size_t)(lane & 15) * Dc + k + 32, 0, 1);

    v16bf a[4];
    #pragma unroll
    for (int mi = 0; mi < 4; ++mi)
      a[mi] = load_frag_a(Abase + (size_t)(mi * 16) * Dc, Dc, k);

    #pragma unroll
    for (int ci = 0; ci < 4; ++ci) {
      const v16bf b = load_frag_b(Wbase + (size_t)(ci * 16) * Dc, Dc, k);
      #pragma unroll
      for (int mi = 0; mi < 4; ++mi)
        acc[mi][ci] = wmma_bf16(a[mi], b, acc[mi][ci]);
    }
  }

  #pragma unroll
  for (int ci = 0; ci < 4; ++ci) {
    const int   n  = n0 + ci * 16 + (lane & 15);
    const float bv = bias[n];
    const int   h  = n / HDc, hd = n % HDc;
    #pragma unroll
    for (int mi = 0; mi < 4; ++mi) {
      #pragma unroll
      for (int r = 0; r < 8; ++r) {
        const int   m   = m0 + mi * 16 + r + (lh << 3);
        const float val = acc[mi][ci][r] + bv;
        const int   b   = m / Sc, s = m % Sc;
        if (mode == 3) {
          out_f[(size_t)m * Dc + n] = val;
        } else if (mode == 2) {
          out_bf[(((size_t)b * Hc + h) * HDc + hd) * Sc + s] = (__bf16)val;
        } else {
          out_bf[(((size_t)b * Hc + h) * Sc + s) * HDc + hd] = (__bf16)val;
        }
      }
    }
  }
}

// ---------------------------------------------------------------------------
// Fused flash attention. One wave per (b*h, 32-query tile): two 16-query
// sub-tiles share every K-tile and V-tile fragment (32 FLOP/byte).
// Scores computed transposed: St[key,q] = K_tile(16x64) x Q^T -> the St
// D-fragment registers are exactly the A-fragment layout of P for P x V.
// ---------------------------------------------------------------------------
__global__ void attn_fa(const __bf16* __restrict__ Q, const __bf16* __restrict__ Kb,
                        const __bf16* __restrict__ Vt, __bf16* __restrict__ Ctx) {
  const int lane   = threadIdx.x & 31;
  const int lh     = lane >> 4;               // lane half
  const int wave   = blockIdx.x * (blockDim.x >> 5) + (threadIdx.x >> 5);
  const int qtiles = Sc / 32;                 // 64 double-tiles
  const int bh     = wave / qtiles;
  const int qt     = wave % qtiles;
  if (bh >= Bc * Hc) return;

  const __bf16* Qh = Q  + (size_t)bh * Sc * HDc;
  const __bf16* Kh = Kb + (size_t)bh * Sc * HDc;
  const __bf16* Vh = Vt + (size_t)bh * HDc * Sc;

  // Q fragments for the two query sub-tiles, [j][lo/hi over hd]
  v16bf bq[2][2];
  #pragma unroll
  for (int j = 0; j < 2; ++j) {
    const __bf16* qb = Qh + (size_t)(qt * 32 + j * 16) * HDc;
    bq[j][0] = load_frag_b(qb, HDc, 0);
    bq[j][1] = load_frag_b(qb, HDc, 32);
  }

  v8f acc[2][4];
  #pragma unroll
  for (int j = 0; j < 2; ++j)
    #pragma unroll
    for (int c = 0; c < 4; ++c)
      #pragma unroll
      for (int r = 0; r < 8; ++r) acc[j][c][r] = 0.f;
  float m_run[2] = {-3.0e38f, -3.0e38f};
  float l_run[2] = {0.f, 0.f};

  v16bf pA[2];

  for (int kt = 0; kt < Sc / 32; ++kt) {
    const __bf16* Kt0 = Kh + (size_t)kt * 32 * HDc;
    v16bf ak[2][2];                           // [key block][lo/hi over hd]
    #pragma unroll
    for (int kb = 0; kb < 2; ++kb) {
      ak[kb][0] = load_frag_a(Kt0 + (size_t)(kb * 16) * HDc, HDc, 0);
      ak[kb][1] = load_frag_a(Kt0 + (size_t)(kb * 16) * HDc, HDc, 32);
    }

    #pragma unroll
    for (int j = 0; j < 2; ++j) {
      v8f st0, st1;
      #pragma unroll
      for (int r = 0; r < 8; ++r) { st0[r] = 0.f; st1[r] = 0.f; }
      st0 = wmma_bf16(ak[0][0], bq[j][0], st0);
      st0 = wmma_bf16(ak[0][1], bq[j][1], st0);
      st1 = wmma_bf16(ak[1][0], bq[j][0], st1);
      st1 = wmma_bf16(ak[1][1], bq[j][1], st1);

      // scale by 1/sqrt(HD); tile max per query (q = lane & 15)
      float s0[8], s1[8];
      float tm = -3.0e38f;
      #pragma unroll
      for (int r = 0; r < 8; ++r) {
        s0[r] = st0[r] * 0.125f;
        s1[r] = st1[r] * 0.125f;
        tm = fmaxf(tm, fmaxf(s0[r], s1[r]));
      }
      tm = fmaxf(tm, __shfl_xor(tm, 16, 32));
      const float mnew  = fmaxf(m_run[j], tm);
      const float alpha = __expf(m_run[j] - mnew);

      float rsum = 0.f;
      #pragma unroll
      for (int r = 0; r < 8; ++r) {
        s0[r] = __expf(s0[r] - mnew);
        s1[r] = __expf(s1[r] - mnew);
        rsum += s0[r] + s1[r];
      }
      rsum += __shfl_xor(rsum, 16, 32);
      l_run[j] = l_run[j] * alpha + rsum;
      m_run[j] = mnew;

      // redistribute per-query alpha to the ctx-accumulator row mapping
      float arow[8];
      #pragma unroll
      for (int r = 0; r < 8; ++r) arow[r] = __shfl(alpha, r + (lh << 3), 32);
      #pragma unroll
      for (int c = 0; c < 4; ++c)
        #pragma unroll
        for (int r = 0; r < 8; ++r) acc[j][c][r] *= arow[r];

      // P fragment: St registers already match A-fragment layout (16q x 32k)
      #pragma unroll
      for (int r = 0; r < 8; ++r) {
        pA[j][r]     = (__bf16)s0[r];
        pA[j][8 + r] = (__bf16)s1[r];
      }
    }

    // ctx accumulation: V fragments shared by both query sub-tiles
    #pragma unroll
    for (int c = 0; c < 4; ++c) {
      const v16bf vb = load_frag_b(Vh + (size_t)(c * 16) * Sc, Sc, kt * 32);
      acc[0][c] = wmma_bf16(pA[0], vb, acc[0][c]);
      acc[1][c] = wmma_bf16(pA[1], vb, acc[1][c]);
    }
  }

  // normalize and store ctx as [b, s, h*HD + hd] (bf16)
  const int b = bh / Hc, h = bh % Hc;
  #pragma unroll
  for (int j = 0; j < 2; ++j) {
    const float linv = 1.f / l_run[j];
    float lrow[8];
    #pragma unroll
    for (int r = 0; r < 8; ++r) lrow[r] = __shfl(linv, r + (lh << 3), 32);
    #pragma unroll
    for (int c = 0; c < 4; ++c) {
      const int hd = c * 16 + (lane & 15);
      #pragma unroll
      for (int r = 0; r < 8; ++r) {
        const int s = qt * 32 + j * 16 + r + (lh << 3);
        Ctx[((size_t)b * Sc + s) * Dc + h * HDc + hd] =
            (__bf16)(acc[j][c][r] * lrow[r]);
      }
    }
  }
}

// ---------------------------------------------------------------------------
extern "C" void kernel_launch(void* const* d_in, const int* in_sizes, int n_in,
                              void* d_out, int out_size, void* d_ws, size_t ws_size,
                              hipStream_t stream) {
  (void)in_sizes; (void)n_in; (void)out_size; (void)ws_size;
  const float* hs = (const float*)d_in[0];
  const float* wq = (const float*)d_in[1];
  const float* bq = (const float*)d_in[2];
  const float* wk = (const float*)d_in[3];
  const float* bk = (const float*)d_in[4];
  const float* wv = (const float*)d_in[5];
  const float* bv = (const float*)d_in[6];
  const float* wo = (const float*)d_in[7];
  const float* bo = (const float*)d_in[8];
  float* out = (float*)d_out;

  char* ws = (char*)d_ws;
  size_t off = 0;
  __bf16* Xbf = (__bf16*)(ws + off); off += (size_t)Mg * Dc * 2;   // 8 MB
  __bf16* Wqb = (__bf16*)(ws + off); off += (size_t)Dc * Dc * 2;   // 2 MB
  __bf16* Wkb = (__bf16*)(ws + off); off += (size_t)Dc * Dc * 2;
  __bf16* Wvb = (__bf16*)(ws + off); off += (size_t)Dc * Dc * 2;
  __bf16* Wob = (__bf16*)(ws + off); off += (size_t)Dc * Dc * 2;
  __bf16* Qbf = (__bf16*)(ws + off); off += (size_t)Mg * Dc * 2;   // [b,h,s,hd]
  __bf16* Kbf = (__bf16*)(ws + off); off += (size_t)Mg * Dc * 2;   // [b,h,s,hd]
  __bf16* Vtb = (__bf16*)(ws + off); off += (size_t)Mg * Dc * 2;   // [b,h,hd,s]
  __bf16* Ctx = (__bf16*)(ws + off); off += (size_t)Mg * Dc * 2;   // [b,s,d]

  // 1) convert fp32 -> bf16
  {
    const int nx4 = Mg * Dc / 4;                        // 1,048,576
    cvt_f32_bf16<<<dim3((nx4 + 255) / 256), dim3(256), 0, stream>>>(hs, Xbf, nx4);
    const int nw4 = Dc * Dc / 4;                        // 262,144
    dim3 g((nw4 + 255) / 256), t(256);
    cvt_f32_bf16<<<g, t, 0, stream>>>(wq, Wqb, nw4);
    cvt_f32_bf16<<<g, t, 0, stream>>>(wk, Wkb, nw4);
    cvt_f32_bf16<<<g, t, 0, stream>>>(wv, Wvb, nw4);
    cvt_f32_bf16<<<g, t, 0, stream>>>(wo, Wob, nw4);
  }

  // 2) Q/K/V projections: (Mg/64)*(Dc/64) = 1024 waves -> 256 blocks x 128 thr
  const dim3 gg(256), gt(128);
  gemm_nt<<<gg, gt, 0, stream>>>(Xbf, Wqb, bq, Qbf, nullptr, 0);
  gemm_nt<<<gg, gt, 0, stream>>>(Xbf, Wkb, bk, Kbf, nullptr, 0);
  gemm_nt<<<gg, gt, 0, stream>>>(Xbf, Wvb, bv, Vtb, nullptr, 2);

  // 3) fused attention: B*H*(S/32) = 2048 waves -> 512 blocks x 128 thr
  attn_fa<<<dim3(512), dim3(128), 0, stream>>>(Qbf, Kbf, Vtb, Ctx);

  // 4) output projection (fp32 result straight to d_out)
  gemm_nt<<<gg, gt, 0, stream>>>(Ctx, Wob, bo, nullptr, out, 3);
}
